// MBN_Model_11725260718355
// MI455X (gfx1250) — matrix operand order repaired and checked by reading, not compiled, not run on hardware
//
#include <hip/hip_runtime.h>
#include <hip/hip_bf16.h>

// ---------------------------------------------------------------------------
// MBN next-basket model for MI455X (gfx1250, wave32, WMMA).
//  - bf16 WMMA (f32 accum) for all GEMMs (GRU gates + score_new).
//  - score_new register-blocked 32x80 per wave: 10 WMMA per 14 b128 loads.
//  - score_copy computed sparsely (C=20 items/row) + bulk zero-fill with
//    non-temporal stores: output write traffic (460MB) dominates.
// ---------------------------------------------------------------------------

#define DEV __device__ __forceinline__

typedef __bf16 bf16;
typedef __attribute__((ext_vector_type(16))) __bf16 bf16x16;
typedef __attribute__((ext_vector_type(8)))  __bf16 bf16x8;
typedef __attribute__((ext_vector_type(8)))  float  f32x8;
typedef __attribute__((ext_vector_type(4)))  float  f32x4;

// problem dims
constexpr int cT = 2, cB = 64, cS = 30, cK = 12, cC = 20;
constexpr int cN = 20000, cDI = 128, cDH = 256, cDIN = 384, cG = 768;

// ---------------- workspace layout (bytes) ----------------
constexpr size_t al(size_t x) { return (x + 255) & ~(size_t)255; }
constexpr size_t OFF_EBASK = 0;                                          // bf16 [S,T,B,128]
constexpr size_t OFF_CMASK = al(OFF_EBASK + (size_t)cS*cT*cB*cDI*2);     // f32  [S,T,B]
constexpr size_t OFF_WIH   = al(OFF_CMASK + (size_t)cS*cT*cB*4);         // bf16 [T,768,384]
constexpr size_t OFF_WHH   = al(OFF_WIH  + (size_t)cT*cG*cDIN*2);        // bf16 [T,768,256]
constexpr size_t OFF_MWIH  = al(OFF_WHH  + (size_t)cT*cG*cDH*2);         // bf16 [768,512]
constexpr size_t OFF_MWHH  = al(OFF_MWIH + (size_t)cG*(cT*cDH)*2);       // bf16 [768,256]
constexpr size_t OFF_WNEW  = al(OFF_MWHH + (size_t)cG*cDH*2);            // bf16 [20000,256]
constexpr size_t OFF_HTF   = al(OFF_WNEW + (size_t)cN*cDH*2);            // f32  2x[T,B,256]
constexpr size_t OFF_HTB   = OFF_HTF + (size_t)2*cT*cB*cDH*4;            // bf16 2x[T,B,256]
constexpr size_t OFF_HMF   = OFF_HTB + (size_t)2*cT*cB*cDH*2;            // f32  2x[B,256]
constexpr size_t OFF_HMB   = OFF_HMF + (size_t)2*cB*cDH*4;               // bf16 2x[B,256]
constexpr size_t OFF_HEND  = OFF_HMB + (size_t)2*cB*cDH*2;
constexpr size_t OFF_HSEQ  = al(OFF_HEND);                               // f32  [S,T,B,256]
constexpr size_t OFF_HMSEQ = al(OFF_HSEQ + (size_t)cS*cT*cB*cDH*4);      // bf16 [B*S,256]

// ---------------- WMMA fragment helpers (ISA 7.12.2 layouts) ----------------
// A 16x32 bf16 : lane holds row m = m0+(lane&15); K chunks {kk..kk+7, kk+16..kk+23},
//                kk = k0 + (lane&16 ? 8 : 0)  -> two contiguous 16B loads.
DEV bf16x16 load_tile_a(const bf16* base, int lda, int m0, int k0, int lane) {
  const int m  = m0 + (lane & 15);
  const int kk = k0 + ((lane & 16) ? 8 : 0);
  const bf16* p = base + (size_t)m * lda + kk;
  bf16x8 lo = *reinterpret_cast<const bf16x8*>(p);
  bf16x8 hi = *reinterpret_cast<const bf16x8*>(p + 16);
  bf16x16 r;
#pragma unroll
  for (int i = 0; i < 8; ++i) { r[i] = lo[i]; r[i + 8] = hi[i]; }
  return r;
}
// B 32x16 bf16 from row-major weights W[out][in] (row n == column n of B):
// lane holds col n = n0+(lane&15); K chunks {kk..kk+7, kk+16..kk+23}, kk = k0 + (lane&16 ? 8 : 0).
DEV bf16x16 load_tile_b(const bf16* base, int ldw, int n0, int k0, int lane) {
  const int n  = n0 + (lane & 15);
  const int kk = k0 + ((lane & 16) ? 8 : 0);
  const bf16* p = base + (size_t)n * ldw + kk;
  bf16x8 lo = *reinterpret_cast<const bf16x8*>(p);
  bf16x8 hi = *reinterpret_cast<const bf16x8*>(p + 16);
  bf16x16 r;
#pragma unroll
  for (int i = 0; i < 8; ++i) { r[i] = lo[i]; r[i + 8] = hi[i]; }
  return r;
}
DEV f32x8 wmma_bf16(bf16x16 a, bf16x16 b, f32x8 c) {
  return __builtin_amdgcn_wmma_f32_16x16x32_bf16(false, a, false, b, (short)0, c,
                                                 false, false);
}
DEV float sigmoidf(float x) { return 1.0f / (1.0f + __expf(-x)); }

// ---------------- utility kernels ----------------
__global__ void convert_f32_bf16_kernel(const float* __restrict__ src,
                                        bf16* __restrict__ dst, long long n) {
  long long i = (long long)blockIdx.x * blockDim.x + threadIdx.x;
  long long stride = (long long)gridDim.x * blockDim.x;
  for (; i < n; i += stride) dst[i] = (bf16)src[i];
}

__global__ void zero_f4_kernel(f32x4* __restrict__ p, long long n4) {
  long long i = (long long)blockIdx.x * blockDim.x + threadIdx.x;
  long long stride = (long long)gridDim.x * blockDim.x;
  f32x4 z = {0.f, 0.f, 0.f, 0.f};
  for (; i < n4; i += stride) __builtin_nontemporal_store(z, p + i);
}

// ---------------- phase 1: basket embedding average pool ----------------
// grid = T*B*S blocks (pos = ((t*B)+b)*S+s), 128 threads (feature dim)
__global__ void pool_kernel(const int* __restrict__ items,
                            const float* __restrict__ emb,
                            bf16* __restrict__ ebask, float* __restrict__ cmask) {
  int pos = blockIdx.x, tid = threadIdx.x;
  int s = pos % cS; int tmp = pos / cS;
  int b = tmp % cB; int t = tmp / cB;
  const int* iv = items + ((size_t)(t * cB + b) * cS + s) * cK;
  float sum = 0.f; int nv = 0;
#pragma unroll
  for (int k = 0; k < cK; ++k) {
    int it = iv[k];                    // padding_idx==0 -> excluded & zero emb
    if (it > 0) { ++nv; sum += emb[(size_t)it * cDI + tid]; }
  }
  float den = (nv > 0) ? (float)nv : 1.0f;
  ebask[((size_t)(s * cT + t) * cB + b) * cDI + tid] = (bf16)(sum / den);
  if (tid == 0) cmask[(size_t)(s * cT + t) * cB + b] = (nv > 0) ? 1.0f : 0.0f;
}

// ---------------- phase 2a: per-task GRU step (one launch per s) ----------------
// 16 blocks x 256 threads = 128 waves; wave tile = (t, m0, c0), c0 in hidden dim.
__global__ void task_gru_kernel(const bf16* __restrict__ x,       // [T,B,128]
                                const bf16* __restrict__ hmeta_b, // [B,256] (read)
                                const bf16* __restrict__ htask_b, // [T,B,256] (read)
                                const float* __restrict__ htask_f,// [T,B,256] (read)
                                const bf16* __restrict__ WihB,    // [T,768,384]
                                const bf16* __restrict__ WhhB,    // [T,768,256]
                                const float* __restrict__ bih,    // [T,768]
                                const float* __restrict__ bhh,    // [T,768]
                                const float* __restrict__ cmask,  // [T,B]
                                float* __restrict__ htask_f_w,    // [T,B,256]
                                bf16* __restrict__ htask_b_w,     // [T,B,256]
                                float* __restrict__ hseq_w) {     // [T,B,256]
  int wid  = blockIdx.x * 8 + (threadIdx.x >> 5);
  int lane = threadIdx.x & 31;
  int t    = wid >> 6;
  int rem  = wid & 63;
  int m0   = (rem >> 4) * 16;
  int c0   = (rem & 15) * 16;

  f32x8 ai[3] = {};   // r,z,n of gi
  f32x8 ah[3] = {};   // r,z,n of gh
  const bf16* W1 = WihB + (size_t)t * cG * cDIN;
  const bf16* W2 = WhhB + (size_t)t * cG * cDH;

  // gi part 1: x(128) block of cell_in
  const bf16* A1 = x + (size_t)t * cB * cDI;
#pragma unroll
  for (int k0 = 0; k0 < cDI; k0 += 32) {
    bf16x16 a = load_tile_a(A1, cDI, m0, k0, lane);
#pragma unroll
    for (int g = 0; g < 3; ++g)
      ai[g] = wmma_bf16(a, load_tile_b(W1, cDIN, g * cDH + c0, k0, lane), ai[g]);
  }
  // gi part 2: h_meta(256) block of cell_in (concat handled by K-offset)
#pragma unroll
  for (int k0 = 0; k0 < cDH; k0 += 32) {
    bf16x16 a = load_tile_a(hmeta_b, cDH, m0, k0, lane);
#pragma unroll
    for (int g = 0; g < 3; ++g)
      ai[g] = wmma_bf16(a, load_tile_b(W1, cDIN, g * cDH + c0, cDI + k0, lane), ai[g]);
  }
  // gh: h_task @ Whh^T
  const bf16* A3 = htask_b + (size_t)t * cB * cDH;
#pragma unroll
  for (int k0 = 0; k0 < cDH; k0 += 32) {
    bf16x16 a = load_tile_a(A3, cDH, m0, k0, lane);
#pragma unroll
    for (int g = 0; g < 3; ++g)
      ah[g] = wmma_bf16(a, load_tile_b(W2, cDH, g * cDH + c0, k0, lane), ah[g]);
  }

  int col = c0 + (lane & 15);
  float bi_r = bih[t * cG + col],            bh_r = bhh[t * cG + col];
  float bi_z = bih[t * cG + cDH + col],      bh_z = bhh[t * cG + cDH + col];
  float bi_n = bih[t * cG + 2 * cDH + col],  bh_n = bhh[t * cG + 2 * cDH + col];
#pragma unroll
  for (int v = 0; v < 8; ++v) {
    int m = m0 + v + ((lane & 16) ? 8 : 0);
    size_t idx = ((size_t)t * cB + m) * cDH + col;
    float r  = sigmoidf(ai[0][v] + bi_r + ah[0][v] + bh_r);
    float z  = sigmoidf(ai[1][v] + bi_z + ah[1][v] + bh_z);
    float nn = tanhf(ai[2][v] + bi_n + r * (ah[2][v] + bh_n));
    float hp = htask_f[idx];
    float hn = (1.0f - z) * nn + z * hp;
    float mk = cmask[t * cB + m];
    float h  = hn * mk + hp * (1.0f - mk);   // freeze on empty basket
    htask_f_w[idx] = h;
    htask_b_w[idx] = (bf16)h;
    hseq_w[idx]    = h;
  }
}

// ---------------- phase 2b: meta GRU step ----------------
// 8 blocks x 256 threads = 64 waves.
__global__ void meta_gru_kernel(const bf16* __restrict__ htask_b, // [T,B,256] (updated)
                                const bf16* __restrict__ hmeta_b, // [B,256] (read)
                                const float* __restrict__ hmeta_f,// [B,256] (read)
                                const bf16* __restrict__ mWihB,   // [768,512]
                                const bf16* __restrict__ mWhhB,   // [768,256]
                                const float* __restrict__ mbih,
                                const float* __restrict__ mbhh,
                                float* __restrict__ hmeta_f_w,
                                bf16* __restrict__ hmeta_b_w,
                                bf16* __restrict__ hm_rec,        // [B*S,256]
                                int s) {
  int wid  = blockIdx.x * 8 + (threadIdx.x >> 5);
  int lane = threadIdx.x & 31;
  int m0   = (wid >> 4) * 16;
  int c0   = (wid & 15) * 16;

  f32x8 ai[3] = {}, ah[3] = {};
  // gi: meta_in = [h_task[t=0] | h_task[t=1]]  (K split over the concat)
#pragma unroll
  for (int t = 0; t < cT; ++t) {
    const bf16* A = htask_b + (size_t)t * cB * cDH;
#pragma unroll
    for (int k0 = 0; k0 < cDH; k0 += 32) {
      bf16x16 a = load_tile_a(A, cDH, m0, k0, lane);
#pragma unroll
      for (int g = 0; g < 3; ++g)
        ai[g] = wmma_bf16(a, load_tile_b(mWihB, cT * cDH, g * cDH + c0,
                                         t * cDH + k0, lane), ai[g]);
    }
  }
  // gh
#pragma unroll
  for (int k0 = 0; k0 < cDH; k0 += 32) {
    bf16x16 a = load_tile_a(hmeta_b, cDH, m0, k0, lane);
#pragma unroll
    for (int g = 0; g < 3; ++g)
      ah[g] = wmma_bf16(a, load_tile_b(mWhhB, cDH, g * cDH + c0, k0, lane), ah[g]);
  }

  int col = c0 + (lane & 15);
  float bi_r = mbih[col],            bh_r = mbhh[col];
  float bi_z = mbih[cDH + col],      bh_z = mbhh[cDH + col];
  float bi_n = mbih[2 * cDH + col],  bh_n = mbhh[2 * cDH + col];
#pragma unroll
  for (int v = 0; v < 8; ++v) {
    int m = m0 + v + ((lane & 16) ? 8 : 0);
    size_t idx = (size_t)m * cDH + col;
    float r  = sigmoidf(ai[0][v] + bi_r + ah[0][v] + bh_r);
    float z  = sigmoidf(ai[1][v] + bi_z + ah[1][v] + bh_z);
    float nn = tanhf(ai[2][v] + bi_n + r * (ah[2][v] + bh_n));
    float hp = hmeta_f[idx];
    float h  = (1.0f - z) * nn + z * hp;
    hmeta_f_w[idx] = h;
    hmeta_b_w[idx] = (bf16)h;
    hm_rec[((size_t)m * cS + s) * cDH + col] = (bf16)h;   // row b*S+s for GEMM A
  }
}

// ---------------- phase 3a: score_new dense GEMM [1920,256]x[256,20000] ----------------
// Register-blocked: each wave computes a 32x80 strip (2 M-tiles x 5 N-tiles).
// Per k-step: 2 A-frags + 5 B-frags feed 10 WMMAs (1.4 b128 loads per WMMA).
// Grid exact: (1920/32)*(20000/80) = 15000 waves = 1875 blocks x 8 waves.
constexpr int SN_NT = 5;                        // N tiles per wave
constexpr int SN_NSTRIPS = cN / (16 * SN_NT);   // 250
__global__ void score_new_kernel(const bf16* __restrict__ hm,    // [1920,256]
                                 const bf16* __restrict__ WnewB, // [20000,256]
                                 const float* __restrict__ bnew,
                                 float* __restrict__ out) {      // slice 0: [1920,20000]
  int wid  = blockIdx.x * 8 + (threadIdx.x >> 5);
  int lane = threadIdx.x & 31;
  int m0   = (wid / SN_NSTRIPS) * 32;
  int n0   = (wid % SN_NSTRIPS) * (16 * SN_NT);

  f32x8 acc0[SN_NT] = {};
  f32x8 acc1[SN_NT] = {};
#pragma unroll
  for (int k0 = 0; k0 < cDH; k0 += 32) {
    bf16x16 a0 = load_tile_a(hm, cDH, m0, k0, lane);
    bf16x16 a1 = load_tile_a(hm, cDH, m0 + 16, k0, lane);
#pragma unroll
    for (int j = 0; j < SN_NT; ++j) {
      bf16x16 b = load_tile_b(WnewB, cDH, n0 + j * 16, k0, lane);
      acc0[j] = wmma_bf16(a0, b, acc0[j]);
      acc1[j] = wmma_bf16(a1, b, acc1[j]);
    }
  }
#pragma unroll
  for (int j = 0; j < SN_NT; ++j) {
    int n = n0 + j * 16 + (lane & 15);
    float bn = bnew[n];
#pragma unroll
    for (int v = 0; v < 8; ++v) {
      int r0 = m0 + v + ((lane & 16) ? 8 : 0);
      __builtin_nontemporal_store(acc0[j][v] + bn, &out[(size_t)r0 * cN + n]);
      __builtin_nontemporal_store(acc1[j][v] + bn, &out[(size_t)(r0 + 16) * cN + n]);
    }
  }
}

// ---------------- phase 3b: sparse copy scores ----------------
// one wave per (t,b,s,c): 76800 waves = 9600 blocks x 8 waves.
__global__ void score_copy_kernel(const int* __restrict__ copyv,  // [T,B,S,C]
                                  const float* __restrict__ hseq, // [S,T,B,256]
                                  const float* __restrict__ Wcopy,// [T,20000,256]
                                  const float* __restrict__ bcopy,// [T,20000]
                                  float* __restrict__ out) {
  int wid  = blockIdx.x * 8 + (threadIdx.x >> 5);
  int lane = threadIdx.x & 31;
  int c = wid % cC; int tmp = wid / cC;
  int s = tmp % cS; tmp /= cS;
  int b = tmp % cB; int t = tmp / cB;
  int item = copyv[((size_t)(t * cB + b) * cS + s) * cC + c];

  const f32x4* h4 = (const f32x4*)(hseq + ((size_t)(s * cT + t) * cB + b) * cDH);
  const f32x4* w4 = (const f32x4*)(Wcopy + ((size_t)t * cN + item) * cDH);
  float sum = 0.f;
#pragma unroll
  for (int i = 0; i < 2; ++i) {                     // 256 = 32 lanes * 2 * 4
    f32x4 a = h4[lane * 2 + i], w = w4[lane * 2 + i];
    sum += a.x * w.x + a.y * w.y + a.z * w.z + a.w * w.w;
  }
#pragma unroll
  for (int off = 16; off > 0; off >>= 1) sum += __shfl_xor(sum, off, 32);
  if (lane == 0)
    out[(size_t)(1 + t) * cB * cS * cN + ((size_t)b * cS + s) * cN + item] =
        sum + bcopy[(size_t)t * cN + item];
}

// ---------------- host driver ----------------
extern "C" void kernel_launch(void* const* d_in, const int* in_sizes, int n_in,
                              void* d_out, int out_size, void* d_ws, size_t ws_size,
                              hipStream_t stream) {
  (void)in_sizes; (void)n_in; (void)out_size; (void)ws_size;
  const int*   item_vectors = (const int*)d_in[1];
  const int*   copy_vectors = (const int*)d_in[2];
  const float* emb   = (const float*)d_in[3];
  const float* Wih   = (const float*)d_in[4];
  const float* Whh   = (const float*)d_in[5];
  const float* bih   = (const float*)d_in[6];
  const float* bhh   = (const float*)d_in[7];
  const float* mWih  = (const float*)d_in[8];
  const float* mWhh  = (const float*)d_in[9];
  const float* mbih  = (const float*)d_in[10];
  const float* mbhh  = (const float*)d_in[11];
  const float* Wnew  = (const float*)d_in[12];
  const float* bnew  = (const float*)d_in[13];
  const float* Wcopy = (const float*)d_in[14];
  const float* bcopy = (const float*)d_in[15];
  float* out = (float*)d_out;
  char*  w   = (char*)d_ws;

  bf16*  ebask = (bf16*) (w + OFF_EBASK);
  float* cmask = (float*)(w + OFF_CMASK);
  bf16*  WihB  = (bf16*) (w + OFF_WIH);
  bf16*  WhhB  = (bf16*) (w + OFF_WHH);
  bf16*  mWihB = (bf16*) (w + OFF_MWIH);
  bf16*  mWhhB = (bf16*) (w + OFF_MWHH);
  bf16*  WnewB = (bf16*) (w + OFF_WNEW);
  float* hTf   = (float*)(w + OFF_HTF);
  bf16*  hTb   = (bf16*) (w + OFF_HTB);
  float* hMf   = (float*)(w + OFF_HMF);
  bf16*  hMb   = (bf16*) (w + OFF_HMB);
  float* hseq  = (float*)(w + OFF_HSEQ);
  bf16*  hmseq = (bf16*) (w + OFF_HMSEQ);

  // weight conversions (f32 -> bf16), done every call for determinism
  convert_f32_bf16_kernel<<<512, 256, 0, stream>>>(Wih,  WihB,  (long long)cT * cG * cDIN);
  convert_f32_bf16_kernel<<<512, 256, 0, stream>>>(Whh,  WhhB,  (long long)cT * cG * cDH);
  convert_f32_bf16_kernel<<<512, 256, 0, stream>>>(mWih, mWihB, (long long)cG * cT * cDH);
  convert_f32_bf16_kernel<<<512, 256, 0, stream>>>(mWhh, mWhhB, (long long)cG * cDH);
  convert_f32_bf16_kernel<<<2048, 256, 0, stream>>>(Wnew, WnewB, (long long)cN * cDH);

  // zero hidden-state buffers (h0 == 0)
  zero_f4_kernel<<<256, 256, 0, stream>>>((f32x4*)(w + OFF_HTF),
                                          (long long)((OFF_HEND - OFF_HTF) / 16));
  // basket average pool
  pool_kernel<<<cT * cB * cS, cDI, 0, stream>>>(item_vectors, emb, ebask, cmask);

  // sequential GRU scan (ping-pong hidden buffers)
  const size_t HT = (size_t)cT * cB * cDH, HM = (size_t)cB * cDH;
  for (int s = 0; s < cS; ++s) {
    int r = s & 1, wr = 1 - r;
    task_gru_kernel<<<16, 256, 0, stream>>>(
        ebask + (size_t)s * cT * cB * cDI, hMb + r * HM, hTb + r * HT, hTf + r * HT,
        WihB, WhhB, bih, bhh, cmask + (size_t)s * cT * cB,
        hTf + wr * HT, hTb + wr * HT, hseq + (size_t)s * HT);
    meta_gru_kernel<<<8, 256, 0, stream>>>(
        hTb + wr * HT, hMb + r * HM, hMf + r * HM, mWihB, mWhhB, mbih, mbhh,
        hMf + wr * HM, hMb + wr * HM, hmseq, s);
  }

  // zero the T copy slices of the output, then fill slice 0 densely and
  // slices 1..T sparsely.
  zero_f4_kernel<<<8192, 256, 0, stream>>>(
      (f32x4*)(out + (size_t)cB * cS * cN), (long long)cT * cB * cS * cN / 4);
  score_new_kernel<<<(cB * cS / 32) * SN_NSTRIPS / 8, 256, 0, stream>>>(
      hmseq, WnewB, bnew, out);
  score_copy_kernel<<<(cT * cB * cS * cC) / 8, 256, 0, stream>>>(
      copy_vectors, hseq, Wcopy, bcopy, out);
}
